// AttentionTrain_3332894622053
// MI455X (gfx1250) — matrix operand order repaired and checked by reading, not compiled, not run on hardware
//
#include <hip/hip_runtime.h>
#include <hip/hip_bf16.h>

typedef __attribute__((ext_vector_type(16))) __bf16 v16bf;
typedef __attribute__((ext_vector_type(8)))  float  v8f;

// Problem constants (match reference)
#define BATCH 8
#define SEQ   2048
#define HDIM  1024

#define TD      16                 // dec rows per workgroup
#define WAVES   16
#define THREADS (WAVES * 32)
#define EBLK    64                 // enc rows staged per phase-2 block
#define NBLK    (SEQ / EBLK)

// LDS pitches in elements (padded for bank spread; rows stay 16B-aligned)
#define QP  (HDIM + 8)             // bf16 Q tile pitch
#define SP  (SEQ + 8)              // f32 score pitch
#define PP  (SEQ + 8)              // bf16 prob pitch
#define VTP (EBLK + 8)             // bf16 transposed-V pitch (72 -> 144B rows)

#define QBYTES   (TD * QP * 2)
#define PBYTES   (TD * PP * 2)
#define REDBYTES (TD * 32 * 4)
#define MBYTES   (TD * 4)
#define LBYTES   (TD * 4)
#define SBYTES   (TD * SP * 4)          // scratch use #1: scores
#define VTBYTES  (HDIM * VTP * 2)       // scratch use #2: transposed V
#define SCRATCH_OFF (QBYTES + PBYTES + REDBYTES + MBYTES + LBYTES)
#define SCRATCH_BYTES (SBYTES > VTBYTES ? SBYTES : VTBYTES)
#define LDS_TOTAL (SCRATCH_OFF + SCRATCH_BYTES)

__global__ void __launch_bounds__(THREADS, 1)
attn_fused_kernel(const float* __restrict__ enc,
                  const float* __restrict__ dec,
                  float* __restrict__ out)
{
    extern __shared__ char lds[];
    __bf16* Q    = (__bf16*)lds;
    __bf16* P    = (__bf16*)(lds + QBYTES);
    float*  red  = (float*)(lds + QBYTES + PBYTES);
    float*  mrow = (float*)(lds + QBYTES + PBYTES + REDBYTES);
    float*  lrow = (float*)(lds + QBYTES + PBYTES + REDBYTES + MBYTES);
    float*  Sb   = (float*)(lds + SCRATCH_OFF);     // phase 1 / softmax
    __bf16* VT   = (__bf16*)(lds + SCRATCH_OFF);    // phase 2 (aliases Sb)

    const int b  = blockIdx.y;
    const int d0 = blockIdx.x * TD;
    const int tid  = threadIdx.x;
    const int wave = tid >> 5;
    const int lane = tid & 31;

    const float* encB = enc + (size_t)b * SEQ * HDIM;
    const float* decB = dec + (size_t)b * SEQ * HDIM;
    float*       outB = out + (size_t)b * SEQ * (2 * HDIM);

    // ---- Stage Q (dec tile) into LDS as bf16, and copy dec -> out[:, 0:H] ----
    for (int i = tid; i < TD * HDIM / 4; i += THREADS) {
        int r  = i / (HDIM / 4);
        int c4 = i % (HDIM / 4);
        float4 v = ((const float4*)(decB + (size_t)(d0 + r) * HDIM))[c4];
        ((float4*)(outB + (size_t)(d0 + r) * (2 * HDIM)))[c4] = v;
        __bf16* qp = Q + r * QP + c4 * 4;
        qp[0] = (__bf16)v.x; qp[1] = (__bf16)v.y;
        qp[2] = (__bf16)v.z; qp[3] = (__bf16)v.w;
    }
    __syncthreads();

    // Fragment geometry (wave32 WMMA; A and B both use k = 16*(lane/16) + s)
    const int frow  = lane & 15;                 // A row / B col / C col
    const int abase = (lane < 16) ? 0 : 8;       // A K base is 2*abase (k = 16*(lane/16) + s)
    const int klo   = (lane < 16) ? 0 : 16;      // B K-range base
    const int crow  = (lane < 16) ? 0 : 8;       // C row base

    // ================= Phase 1: S = Q * enc^T (per wave: 8 e-tiles) ==========
    for (int et = wave; et < SEQ / 16; et += WAVES) {
        const int e0 = et * 16;
        v8f acc = {};
        const float* ep = encB + (size_t)(e0 + frow) * HDIM + klo;
        #pragma unroll 4
        for (int h0 = 0; h0 < HDIM; h0 += 32) {
            // A: Q[16 x 32] from LDS (two 16B chunks per lane)
            const __bf16* qp = Q + frow * QP + h0;
            v16bf a;
            *(uint4*)&a         = *(const uint4*)(qp + 2 * abase);
            *(((uint4*)&a) + 1) = *(const uint4*)(qp + 8 + 2 * abase);
            // B: enc^T[32 x 16]; lane reads 16 contiguous f32 of one enc row
            float4 f0 = *(const float4*)(ep + h0);
            float4 f1 = *(const float4*)(ep + h0 + 4);
            float4 f2 = *(const float4*)(ep + h0 + 8);
            float4 f3 = *(const float4*)(ep + h0 + 12);
            __builtin_prefetch(ep + h0 + 128, 0, 1);
            v16bf bb;
            bb[0]=(__bf16)f0.x; bb[1]=(__bf16)f0.y; bb[2]=(__bf16)f0.z; bb[3]=(__bf16)f0.w;
            bb[4]=(__bf16)f1.x; bb[5]=(__bf16)f1.y; bb[6]=(__bf16)f1.z; bb[7]=(__bf16)f1.w;
            bb[8]=(__bf16)f2.x; bb[9]=(__bf16)f2.y; bb[10]=(__bf16)f2.z; bb[11]=(__bf16)f2.w;
            bb[12]=(__bf16)f3.x; bb[13]=(__bf16)f3.y; bb[14]=(__bf16)f3.z; bb[15]=(__bf16)f3.w;
            acc = __builtin_amdgcn_wmma_f32_16x16x32_bf16(
                    false, a, false, bb, (short)0, acc, false, false);
        }
        #pragma unroll
        for (int r = 0; r < 8; ++r)
            Sb[(size_t)(crow + r) * SP + e0 + frow] = acc[r];
    }
    __syncthreads();

    // ================= Softmax stats over e (rows d, 32 segments of 64) ======
    {
        const int d  = tid >> 5;
        const int sg = lane;
        const float* srow = Sb + (size_t)d * SP + sg * 64;
        float mx = -1e30f;
        #pragma unroll 8
        for (int j = 0; j < 64; ++j) mx = fmaxf(mx, srow[j]);
        red[d * 32 + sg] = mx;
        __syncthreads();
        if (tid < TD) {
            float m = -1e30f;
            for (int j = 0; j < 32; ++j) m = fmaxf(m, red[tid * 32 + j]);
            mrow[tid] = m;
        }
        __syncthreads();
        const float m = mrow[d];
        __bf16* prow = P + (size_t)d * PP + sg * 64;
        float sum = 0.0f;
        #pragma unroll 8
        for (int j = 0; j < 64; ++j) {
            float ev = __expf(srow[j] - m);
            sum += ev;
            prow[j] = (__bf16)ev;      // un-normalized; 1/l folded into output
        }
        red[d * 32 + sg] = sum;
        __syncthreads();
        if (tid < TD) {
            float s = 0.0f;
            for (int j = 0; j < 32; ++j) s += red[tid * 32 + j];
            lrow[tid] = 1.0f / s;
        }
        // NOTE: scores in Sb are consumed; scratch region is reused as VT below.
    }

    // ================= Phase 2: O = P * enc (wave owns 64 H columns) =========
    const int h0w = wave * 64;
    const int ep2 = lane;          // e-pair index within a 64-row block
    v8f o0 = {}, o1 = {}, o2 = {}, o3 = {};

    for (int blk = 0; blk < NBLK; ++blk) {
        const int e0 = blk * EBLK;
        __syncthreads();   // previous block's VT fully consumed (also closes softmax)

        // ---- Cooperative stage: VT[h][e] = bf16(enc[e0+e][h]), e in [0,64) ----
        {
            const float* r0 = encB + (size_t)(e0 + 2 * ep2) * HDIM;
            const float* r1 = r0 + HDIM;
            #pragma unroll 4
            for (int it = 0; it < 16; ++it) {
                const int h4 = wave * 16 + it;        // float4 index along H
                float4 a0 = ((const float4*)r0)[h4];
                float4 a1 = ((const float4*)r1)[h4];
                const int h = h4 * 4;
                union { __bf16 bh[2]; unsigned u; } w0, w1, w2, w3;
                w0.bh[0] = (__bf16)a0.x; w0.bh[1] = (__bf16)a1.x;
                w1.bh[0] = (__bf16)a0.y; w1.bh[1] = (__bf16)a1.y;
                w2.bh[0] = (__bf16)a0.z; w2.bh[1] = (__bf16)a1.z;
                w3.bh[0] = (__bf16)a0.w; w3.bh[1] = (__bf16)a1.w;
                *(unsigned*)(VT + (size_t)(h + 0) * VTP + 2 * ep2) = w0.u;
                *(unsigned*)(VT + (size_t)(h + 1) * VTP + 2 * ep2) = w1.u;
                *(unsigned*)(VT + (size_t)(h + 2) * VTP + 2 * ep2) = w2.u;
                *(unsigned*)(VT + (size_t)(h + 3) * VTP + 2 * ep2) = w3.u;
            }
        }
        __syncthreads();

        // ---- Consume: 2 K-steps x 4 h-tiles of WMMA from LDS ----------------
        #pragma unroll
        for (int es = 0; es < EBLK; es += 32) {
            const __bf16* pp = P + frow * PP + e0 + es;
            v16bf a;
            *(uint4*)&a         = *(const uint4*)(pp + 2 * abase);
            *(((uint4*)&a) + 1) = *(const uint4*)(pp + 8 + 2 * abase);
            #pragma unroll
            for (int t = 0; t < 4; ++t) {
                const __bf16* vrow = VT + (size_t)(h0w + t * 16 + frow) * VTP + es + klo;
                v16bf bb;
                *(uint4*)&bb         = *(const uint4*)(vrow);
                *(((uint4*)&bb) + 1) = *(const uint4*)(vrow + 8);
                switch (t) {
                    case 0: o0 = __builtin_amdgcn_wmma_f32_16x16x32_bf16(false, a, false, bb, (short)0, o0, false, false); break;
                    case 1: o1 = __builtin_amdgcn_wmma_f32_16x16x32_bf16(false, a, false, bb, (short)0, o1, false, false); break;
                    case 2: o2 = __builtin_amdgcn_wmma_f32_16x16x32_bf16(false, a, false, bb, (short)0, o2, false, false); break;
                    case 3: o3 = __builtin_amdgcn_wmma_f32_16x16x32_bf16(false, a, false, bb, (short)0, o3, false, false); break;
                }
            }
        }
    }

    // ---- Scale by 1/l per row and store context to out[:, H:2H] -------------
    #pragma unroll
    for (int t = 0; t < 4; ++t) {
        const v8f& o = (t == 0) ? o0 : (t == 1) ? o1 : (t == 2) ? o2 : o3;
        #pragma unroll
        for (int r = 0; r < 8; ++r) {
            int M = crow + r;
            float val = o[r] * lrow[M];
            outB[(size_t)(d0 + M) * (2 * HDIM) + HDIM + h0w + t * 16 + frow] = val;
        }
    }
}

extern "C" void kernel_launch(void* const* d_in, const int* in_sizes, int n_in,
                              void* d_out, int out_size, void* d_ws, size_t ws_size,
                              hipStream_t stream) {
    const float* enc = (const float*)d_in[0];   // enc_output [B, S, H]
    const float* dec = (const float*)d_in[1];   // dec_output [B, S, H]
    float* out = (float*)d_out;                 // [B, S, 2H]
    (void)in_sizes; (void)n_in; (void)out_size; (void)d_ws; (void)ws_size;

    dim3 grid(SEQ / TD, BATCH);
    dim3 block(THREADS);
    attn_fused_kernel<<<grid, block, LDS_TOTAL, stream>>>(enc, dec, out);
}
